// EncodeProcessDecode_39917426049693
// MI455X (gfx1250) — matrix operand order repaired and checked by reading, not compiled, not run on hardware
//
#include <hip/hip_runtime.h>
#include <hip/hip_bf16.h>
#include <math.h>

// ---------------------------------------------------------------- constants
#define NNODES 30000
#define NEDGES 150000
#define LAT    128
#define NSTEPS 5
#define TW     5
#define TD     2

typedef __bf16 v16bf __attribute__((ext_vector_type(16)));
typedef __bf16 v8bf  __attribute__((ext_vector_type(8)));
typedef __bf16 v2bf  __attribute__((ext_vector_type(2)));
typedef float  v8f   __attribute__((ext_vector_type(8)));

// ---------------------------------------------------------------- weight pack
// Fragment-native layout: packed[(kb*128 + col)*32 + t] = W[(kb*32 + t)*128 + col]
// so each lane's 16-half B-fragment is two 16-byte chunks, at t = 8h and t = 16 + 8h.
__global__ void pack_w_kernel(const float* __restrict__ src, __bf16* __restrict__ dst,
                              int K, int Kpad) {
    int idx = blockIdx.x * blockDim.x + threadIdx.x;
    int total = Kpad * LAT;
    if (idx >= total) return;
    int t   = idx & 31;
    int col = (idx >> 5) & 127;
    int kb  = idx >> 12;            // 32*128 = 4096 per K-block
    int k   = kb * 32 + t;
    float v = (k < K) ? src[(size_t)k * LAT + col] : 0.0f;
    dst[idx] = (__bf16)v;
}

__global__ void zero_f32_kernel(float4* __restrict__ p, int n4) {
    int idx = blockIdx.x * blockDim.x + threadIdx.x;
    if (idx < n4) p[idx] = make_float4(0.f, 0.f, 0.f, 0.f);
}

// node features: [u(2) | load(1) | node_type(9) | 0-pad] -> bf16 A (stride 32)
__global__ void build_node_feat_kernel(const float* __restrict__ u,
                                       const float* __restrict__ load,
                                       const float* __restrict__ nt,
                                       __bf16* __restrict__ A, int N) {
    int idx = blockIdx.x * blockDim.x + threadIdx.x;
    if (idx >= N * 32) return;
    int n = idx >> 5, k = idx & 31;
    float v = 0.0f;
    if (k < 2)       v = u[n * 2 + k];
    else if (k == 2) v = load[n];
    else if (k < 12) v = nt[n * 9 + (k - 3)];
    A[idx] = (__bf16)v;
}

// edge features: [rel(2) | dist(1) | grad_u(2) | 0-pad] -> bf16 A (stride 32)
__global__ void build_edge_feat_kernel(const float* __restrict__ mesh_pos,
                                       const float* __restrict__ u,
                                       const int* __restrict__ snd,
                                       const int* __restrict__ rcv,
                                       __bf16* __restrict__ A, int E) {
    int e = blockIdx.x * blockDim.x + threadIdx.x;
    if (e >= E) return;
    int s = snd[e], r = rcv[e];
    float rx = mesh_pos[2 * s]     - mesh_pos[2 * r];
    float ry = mesh_pos[2 * s + 1] - mesh_pos[2 * r + 1];
    float ds = sqrtf(rx * rx + ry * ry);
    float g0 = u[2 * s]     - u[2 * r];
    float g1 = u[2 * s + 1] - u[2 * r + 1];
    __bf16* row = A + (size_t)e * 32;
    row[0] = (__bf16)rx; row[1] = (__bf16)ry; row[2] = (__bf16)ds;
    row[3] = (__bf16)g0; row[4] = (__bf16)g1;
    #pragma unroll
    for (int k = 5; k < 32; ++k) row[k] = (__bf16)0.0f;
}

// edge block input: [nl[s] | nl[r] | el] -> bf16 A (stride 384), float2-wide
__global__ void build_edge_A_kernel(const float* __restrict__ nl,
                                    const float* __restrict__ el,
                                    const int* __restrict__ snd,
                                    const int* __restrict__ rcv,
                                    __bf16* __restrict__ A, int E) {
    int idx = blockIdx.x * blockDim.x + threadIdx.x;
    if (idx >= E * 64) return;
    int e = idx >> 6, c = (idx & 63) << 1;
    int s = snd[e], r = rcv[e];
    size_t base = (size_t)e * 384;
    float2 a = *(const float2*)(nl + (size_t)s * LAT + c);
    float2 b = *(const float2*)(nl + (size_t)r * LAT + c);
    float2 d = *(const float2*)(el + (size_t)e * LAT + c);
    *(v2bf*)(A + base + c)       = (v2bf){(__bf16)a.x, (__bf16)a.y};
    *(v2bf*)(A + base + 128 + c) = (v2bf){(__bf16)b.x, (__bf16)b.y};
    *(v2bf*)(A + base + 256 + c) = (v2bf){(__bf16)d.x, (__bf16)d.y};
}

// node block input: [nl | aggr] -> bf16 A (stride 256), float2-wide
__global__ void build_node_A_kernel(const float* __restrict__ nl,
                                    const float* __restrict__ aggr,
                                    __bf16* __restrict__ A, int N) {
    int idx = blockIdx.x * blockDim.x + threadIdx.x;
    if (idx >= N * 64) return;
    int n = idx >> 6, c = (idx & 63) << 1;
    size_t base = (size_t)n * 256;
    float2 a = *(const float2*)(nl   + (size_t)n * LAT + c);
    float2 b = *(const float2*)(aggr + (size_t)n * LAT + c);
    *(v2bf*)(A + base + c)       = (v2bf){(__bf16)a.x, (__bf16)a.y};
    *(v2bf*)(A + base + 128 + c) = (v2bf){(__bf16)b.x, (__bf16)b.y};
}

// scatter-add ne into aggr[receiver], and edge residual el += ne (float2-wide)
__global__ void scatter_resid_kernel(const float* __restrict__ ne,
                                     const int* __restrict__ rcv,
                                     float* __restrict__ aggr,
                                     float* __restrict__ el, int E) {
    int idx = blockIdx.x * blockDim.x + threadIdx.x;
    if (idx >= E * 64) return;
    int e = idx >> 6, c = (idx & 63) << 1;
    size_t base = (size_t)e * LAT + c;
    float2 v = *(const float2*)(ne + base);
    int rr = rcv[e];
    atomicAdd(&aggr[(size_t)rr * LAT + c],     v.x);
    atomicAdd(&aggr[(size_t)rr * LAT + c + 1], v.y);
    float2* pe = (float2*)(el + base);
    float2 cur = *pe;
    cur.x += v.x; cur.y += v.y;
    *pe = cur;
}

// ---------------------------------------------------------------- fused MLP
// two-layer MLP (Linear->ReLU->Linear->ReLU->LayerNorm) over 32-row tiles.
// 8 waves/block; each wave owns a 16-wide slab of the 128 output columns and
// two 16-row accumulators, so every weight B-fragment feeds two WMMAs.
template <int K1>
__global__ __launch_bounds__(256)
void mlp2_wmma_kernel(const __bf16* __restrict__ A,
                      const __bf16* __restrict__ W1p, const float* __restrict__ b1,
                      const __bf16* __restrict__ W2p, const float* __restrict__ b2,
                      const float* __restrict__ gam, const float* __restrict__ bet,
                      const float* __restrict__ resid, float* __restrict__ out,
                      int rows) {
    constexpr int M = 32;
    __shared__ __align__(16) __bf16 sA[M * K1];    // staged A tile
    __shared__ __align__(16) __bf16 sH[M * 128];   // layer-1 output (bf16)
    __shared__ __align__(16) float  sO[M * 128];   // layer-2 output (f32)

    const int tid  = threadIdx.x;
    const int wave = tid >> 5;
    const int lane = tid & 31;
    const int n0   = wave << 4;              // column slab base
    const int nn   = lane & 15;              // M for A-frag / N for B,C,D
    const int hiA  = (lane & 16) ? 8 : 0;    // A-frag K offset (hi half-wave)
    const int hiB  = (lane & 16) ? 16 : 0;   // unused: B-frag K offset is hiA
    const int mOff = (lane & 16) ? 8 : 0;    // C/D row offset (hi half-wave)

    const float bias1 = b1[n0 + nn];
    const float bias2 = b2[n0 + nn];

    const int ntiles = (rows + M - 1) / M;

    for (int tile = blockIdx.x; tile < ntiles; tile += gridDim.x) {
        const int row0  = tile * M;
        const int vrows = min(M, rows - row0);

        // cooperative A-tile stage: flat b128 copy, zero-fill the tail tile
        {
            const uint4* src = (const uint4*)(A + (size_t)row0 * K1);
            uint4* dst = (uint4*)sA;
            const int nvec = (vrows * K1) >> 3;          // 8 halves per uint4
            for (int t = tid; t < (M * K1) >> 3; t += 256)
                dst[t] = (t < nvec) ? src[t] : make_uint4(0u, 0u, 0u, 0u);
        }
        __syncthreads();

        // ---- layer 1: [32 x K1] @ [K1 x 128] ----
        v8f acc0, acc1;
        #pragma unroll
        for (int r = 0; r < 8; ++r) { acc0[r] = bias1; acc1[r] = bias1; }
        #pragma unroll
        for (int k0 = 0; k0 < K1; k0 += 32) {
            const int kb = k0 >> 5;
            v8bf b0 = *(const v8bf*)(W1p + (((size_t)kb * LAT + n0 + nn) << 5) + hiA);
            v8bf b1v = *(const v8bf*)(W1p + (((size_t)kb * LAT + n0 + nn) << 5) + hiA + 16);
            v16bf b = __builtin_shufflevector(b0, b1v, 0, 1, 2, 3, 4, 5, 6, 7,
                                              8, 9, 10, 11, 12, 13, 14, 15);
            v8bf a0 = *(const v8bf*)(sA + nn * K1 + k0 + hiA);
            v8bf a1 = *(const v8bf*)(sA + nn * K1 + k0 + hiA + 16);
            v16bf a = __builtin_shufflevector(a0, a1, 0, 1, 2, 3, 4, 5, 6, 7,
                                              8, 9, 10, 11, 12, 13, 14, 15);
            acc0 = __builtin_amdgcn_wmma_f32_16x16x32_bf16(
                false, a, false, b, (short)0, acc0, false, false);
            v8bf c0 = *(const v8bf*)(sA + (16 + nn) * K1 + k0 + hiA);
            v8bf c1 = *(const v8bf*)(sA + (16 + nn) * K1 + k0 + hiA + 16);
            v16bf a2 = __builtin_shufflevector(c0, c1, 0, 1, 2, 3, 4, 5, 6, 7,
                                               8, 9, 10, 11, 12, 13, 14, 15);
            acc1 = __builtin_amdgcn_wmma_f32_16x16x32_bf16(
                false, a2, false, b, (short)0, acc1, false, false);
        }
        #pragma unroll
        for (int r = 0; r < 8; ++r) {
            float v0 = acc0[r] > 0.0f ? acc0[r] : 0.0f;          // ReLU
            float v1 = acc1[r] > 0.0f ? acc1[r] : 0.0f;
            sH[(r + mOff) * LAT + n0 + nn]        = (__bf16)v0;
            sH[(16 + r + mOff) * LAT + n0 + nn]   = (__bf16)v1;
        }
        __syncthreads();

        // ---- layer 2: [32 x 128] @ [128 x 128] ----
        #pragma unroll
        for (int r = 0; r < 8; ++r) { acc0[r] = bias2; acc1[r] = bias2; }
        #pragma unroll
        for (int k0 = 0; k0 < 128; k0 += 32) {
            const int kb = k0 >> 5;
            v8bf b0 = *(const v8bf*)(W2p + (((size_t)kb * LAT + n0 + nn) << 5) + hiA);
            v8bf b1v = *(const v8bf*)(W2p + (((size_t)kb * LAT + n0 + nn) << 5) + hiA + 16);
            v16bf b = __builtin_shufflevector(b0, b1v, 0, 1, 2, 3, 4, 5, 6, 7,
                                              8, 9, 10, 11, 12, 13, 14, 15);
            v8bf a0 = *(const v8bf*)(sH + nn * LAT + k0 + hiA);
            v8bf a1 = *(const v8bf*)(sH + nn * LAT + k0 + hiA + 16);
            v16bf a = __builtin_shufflevector(a0, a1, 0, 1, 2, 3, 4, 5, 6, 7,
                                              8, 9, 10, 11, 12, 13, 14, 15);
            acc0 = __builtin_amdgcn_wmma_f32_16x16x32_bf16(
                false, a, false, b, (short)0, acc0, false, false);
            v8bf c0 = *(const v8bf*)(sH + (16 + nn) * LAT + k0 + hiA);
            v8bf c1 = *(const v8bf*)(sH + (16 + nn) * LAT + k0 + hiA + 16);
            v16bf a2 = __builtin_shufflevector(c0, c1, 0, 1, 2, 3, 4, 5, 6, 7,
                                               8, 9, 10, 11, 12, 13, 14, 15);
            acc1 = __builtin_amdgcn_wmma_f32_16x16x32_bf16(
                false, a2, false, b, (short)0, acc1, false, false);
        }
        #pragma unroll
        for (int r = 0; r < 8; ++r) {
            float v0 = acc0[r] > 0.0f ? acc0[r] : 0.0f;          // ReLU
            float v1 = acc1[r] > 0.0f ? acc1[r] : 0.0f;
            sO[(r + mOff) * LAT + n0 + nn]      = v0;
            sO[(16 + r + mOff) * LAT + n0 + nn] = v1;
        }
        __syncthreads();

        // ---- LayerNorm over 128 cols per row, + optional residual ----
        #pragma unroll
        for (int rr = 0; rr < 2; ++rr) {
            const int row = (tid >> 4) + rr * 16;    // 16 threads per row
            const int j   = tid & 15;
            float s = 0.0f, sq = 0.0f;
            #pragma unroll
            for (int k = 0; k < 8; ++k) {
                float v = sO[row * LAT + j + 16 * k];
                s  += v;
                sq += v * v;
            }
            #pragma unroll
            for (int off = 8; off >= 1; off >>= 1) {
                s  += __shfl_xor(s,  off, 16);
                sq += __shfl_xor(sq, off, 16);
            }
            float mu   = s  * (1.0f / 128.0f);
            float var  = sq * (1.0f / 128.0f) - mu * mu;
            float rstd = rsqrtf(var + 1e-5f);
            if (row < vrows) {
                size_t obase = (size_t)(row0 + row) * LAT;
                #pragma unroll
                for (int k = 0; k < 8; ++k) {
                    int col = j + 16 * k;
                    float v = sO[row * LAT + col];
                    float y = gam[col] * (v - mu) * rstd + bet[col];
                    if (resid) y += resid[obase + col];
                    out[obase + col] = y;
                }
            }
        }
        __syncthreads();
    }
}

// ---------------------------------------------------------------- decoder
// per-node: 128->8 swish -> 8->10, scaled by dt, laid out [TW, N, TD]
__global__ __launch_bounds__(256)
void decode_kernel(const float* __restrict__ nl,
                   const float* __restrict__ W1, const float* __restrict__ b1,
                   const float* __restrict__ W2, const float* __restrict__ b2,
                   float* __restrict__ out, int N) {
    __shared__ float sW1[128 * 8];
    __shared__ float sW2[8 * 10];
    __shared__ float sb1[8];
    __shared__ float sb2[10];
    for (int t = threadIdx.x; t < 128 * 8; t += 256) sW1[t] = W1[t];
    if (threadIdx.x < 80) sW2[threadIdx.x] = W2[threadIdx.x];
    if (threadIdx.x < 8)  sb1[threadIdx.x] = b1[threadIdx.x];
    if (threadIdx.x < 10) sb2[threadIdx.x] = b2[threadIdx.x];
    __syncthreads();

    int n = blockIdx.x * blockDim.x + threadIdx.x;
    if (n >= N) return;
    const float* xr = nl + (size_t)n * LAT;

    float h[8];
    #pragma unroll
    for (int j = 0; j < 8; ++j) h[j] = sb1[j];
    for (int k = 0; k < 128; ++k) {
        float xv = xr[k];
        #pragma unroll
        for (int j = 0; j < 8; ++j) h[j] += xv * sW1[k * 8 + j];
    }
    #pragma unroll
    for (int j = 0; j < 8; ++j) h[j] = h[j] / (1.0f + expf(-h[j]));   // swish

    #pragma unroll
    for (int j = 0; j < TW * TD; ++j) {
        float d = sb2[j];
        #pragma unroll
        for (int k = 0; k < 8; ++k) d += h[k] * sW2[k * 10 + j];
        int w = j / TD, td = j % TD;
        out[((size_t)w * N + n) * TD + td] = d * (float)(w + 1);
    }
}

// ---------------------------------------------------------------- launch
extern "C" void kernel_launch(void* const* d_in, const int* in_sizes, int n_in,
                              void* d_out, int out_size, void* d_ws, size_t ws_size,
                              hipStream_t stream) {
    (void)in_sizes; (void)n_in; (void)out_size; (void)ws_size;

    const float* mesh_pos  = (const float*)d_in[0];
    const float* node_type = (const float*)d_in[1];
    const float* u         = (const float*)d_in[2];
    const float* load      = (const float*)d_in[3];
    const int*   senders   = (const int*)d_in[4];
    const int*   receivers = (const int*)d_in[5];
    const float* enW1 = (const float*)d_in[6];  const float* enb1 = (const float*)d_in[7];
    const float* enW2 = (const float*)d_in[8];  const float* enb2 = (const float*)d_in[9];
    const float* eng  = (const float*)d_in[10]; const float* enbe = (const float*)d_in[11];
    const float* eeW1 = (const float*)d_in[12]; const float* eeb1 = (const float*)d_in[13];
    const float* eeW2 = (const float*)d_in[14]; const float* eeb2 = (const float*)d_in[15];
    const float* eeg  = (const float*)d_in[16]; const float* eebe = (const float*)d_in[17];
    const float* beW1 = (const float*)d_in[18]; const float* beb1 = (const float*)d_in[19];
    const float* beW2 = (const float*)d_in[20]; const float* beb2 = (const float*)d_in[21];
    const float* beg  = (const float*)d_in[22]; const float* bebe = (const float*)d_in[23];
    const float* bnW1 = (const float*)d_in[24]; const float* bnb1 = (const float*)d_in[25];
    const float* bnW2 = (const float*)d_in[26]; const float* bnb2 = (const float*)d_in[27];
    const float* bng  = (const float*)d_in[28]; const float* bnbe = (const float*)d_in[29];
    const float* dW1  = (const float*)d_in[30]; const float* db1  = (const float*)d_in[31];
    const float* dW2  = (const float*)d_in[32]; const float* db2  = (const float*)d_in[33];
    float* out = (float*)d_out;

    // ---- carve workspace ----
    char* ws = (char*)d_ws;
    size_t off = 0;
    auto carve = [&](size_t bytes) -> char* {
        char* p = ws + off;
        off = (off + bytes + 255) & ~(size_t)255;
        return p;
    };
    __bf16* wEnW1 = (__bf16*)carve((size_t)32 * 128 * 2);
    __bf16* wEnW2 = (__bf16*)carve((size_t)128 * 128 * 2);
    __bf16* wEeW1 = (__bf16*)carve((size_t)32 * 128 * 2);
    __bf16* wEeW2 = (__bf16*)carve((size_t)128 * 128 * 2);
    __bf16* wBeW1 = (__bf16*)carve((size_t)NSTEPS * 384 * 128 * 2);
    __bf16* wBeW2 = (__bf16*)carve((size_t)NSTEPS * 128 * 128 * 2);
    __bf16* wBnW1 = (__bf16*)carve((size_t)NSTEPS * 256 * 128 * 2);
    __bf16* wBnW2 = (__bf16*)carve((size_t)NSTEPS * 128 * 128 * 2);
    float*  nl    = (float*)carve((size_t)NNODES * LAT * 4);
    float*  el    = (float*)carve((size_t)NEDGES * LAT * 4);
    float*  ne    = (float*)carve((size_t)NEDGES * LAT * 4);
    float*  aggr  = (float*)carve((size_t)NNODES * LAT * 4);
    __bf16* Aedge = (__bf16*)carve(((size_t)NEDGES + 32) * 384 * 2);
    __bf16* Anode = (__bf16*)carve(((size_t)NNODES + 32) * 256 * 2);

    auto packW = [&](const float* src, __bf16* dst, int K, int Kpad) {
        int total = Kpad * LAT;
        pack_w_kernel<<<(total + 255) / 256, 256, 0, stream>>>(src, dst, K, Kpad);
    };

    // ---- pack weights to bf16 fragment layout (K padded to 32) ----
    packW(enW1, wEnW1, 12, 32);
    packW(enW2, wEnW2, 128, 128);
    packW(eeW1, wEeW1, 5, 32);
    packW(eeW2, wEeW2, 128, 128);
    for (int s = 0; s < NSTEPS; ++s) {
        packW(beW1 + (size_t)s * 384 * 128, wBeW1 + (size_t)s * 384 * 128, 384, 384);
        packW(beW2 + (size_t)s * 128 * 128, wBeW2 + (size_t)s * 128 * 128, 128, 128);
        packW(bnW1 + (size_t)s * 256 * 128, wBnW1 + (size_t)s * 256 * 128, 256, 256);
        packW(bnW2 + (size_t)s * 128 * 128, wBnW2 + (size_t)s * 128 * 128, 128, 128);
    }

    const int ntN32 = (NNODES + 31) / 32;   // 938
    const int ntE32 = (NEDGES + 31) / 32;   // 4688
    const int gN = ntN32 < 2048 ? ntN32 : 2048;
    const int gE = ntE32 < 2048 ? ntE32 : 2048;

    // ---- encoders ----
    build_node_feat_kernel<<<(NNODES * 32 + 255) / 256, 256, 0, stream>>>(
        u, load, node_type, Anode, NNODES);
    mlp2_wmma_kernel<32><<<gN, 256, 0, stream>>>(
        Anode, wEnW1, enb1, wEnW2, enb2, eng, enbe, nullptr, nl, NNODES);

    build_edge_feat_kernel<<<(NEDGES + 255) / 256, 256, 0, stream>>>(
        mesh_pos, u, senders, receivers, Aedge, NEDGES);
    mlp2_wmma_kernel<32><<<gE, 256, 0, stream>>>(
        Aedge, wEeW1, eeb1, wEeW2, eeb2, eeg, eebe, nullptr, el, NEDGES);

    // ---- message-passing blocks ----
    for (int s = 0; s < NSTEPS; ++s) {
        build_edge_A_kernel<<<(NEDGES * 64 + 255) / 256, 256, 0, stream>>>(
            nl, el, senders, receivers, Aedge, NEDGES);
        mlp2_wmma_kernel<384><<<gE, 256, 0, stream>>>(
            Aedge,
            wBeW1 + (size_t)s * 384 * 128, beb1 + (size_t)s * 128,
            wBeW2 + (size_t)s * 128 * 128, beb2 + (size_t)s * 128,
            beg + (size_t)s * 128, bebe + (size_t)s * 128,
            nullptr, ne, NEDGES);

        zero_f32_kernel<<<(NNODES * LAT / 4 + 255) / 256, 256, 0, stream>>>(
            (float4*)aggr, NNODES * LAT / 4);
        scatter_resid_kernel<<<(NEDGES * 64 + 255) / 256, 256, 0, stream>>>(
            ne, receivers, aggr, el, NEDGES);

        build_node_A_kernel<<<(NNODES * 64 + 255) / 256, 256, 0, stream>>>(
            nl, aggr, Anode, NNODES);
        mlp2_wmma_kernel<256><<<gN, 256, 0, stream>>>(
            Anode,
            wBnW1 + (size_t)s * 256 * 128, bnb1 + (size_t)s * 128,
            wBnW2 + (size_t)s * 128 * 128, bnb2 + (size_t)s * 128,
            bng + (size_t)s * 128, bnbe + (size_t)s * 128,
            nl, nl, NNODES);   // residual: nl = nl + MLP(node_in)
    }

    // ---- decoder ----
    decode_kernel<<<(NNODES + 255) / 256, 256, 0, stream>>>(
        nl, dW1, db1, dW2, db2, out, NNODES);
}
